// GatheringLoss_68977174774316
// MI455X (gfx1250) — matrix operand order repaired and checked by reading, not compiled, not run on hardware
//
#include <hip/hip_runtime.h>

typedef __attribute__((ext_vector_type(16))) __bf16 v16bf;
typedef __attribute__((ext_vector_type(8)))  float  v8f;

#define C_DIM 128
#define M_KEYS 1024
#define T_ROWS 65536
#define CHUNK_KEYS 128                      // keys staged per LDS chunk (32 KB bf16)
#define N_CHUNKS (M_KEYS / CHUNK_KEYS)      // 8
#define TILES_PER_CHUNK (CHUNK_KEYS / 16)   // 8
#define WAVES_PER_BLOCK 8
#define RTILES 2                            // 16-row q tiles register-blocked per wave
#define ROWS_PER_WAVE (16 * RTILES)         // 32
#define ROWS_PER_BLOCK (WAVES_PER_BLOCK * ROWS_PER_WAVE)  // 256

__global__ __launch_bounds__(256)
void gathering_loss_kernel(const float* __restrict__ q,       // (T, C) trend
                           const float* __restrict__ r,       // (T, C) representation
                           const float* __restrict__ keys,    // (M, C)
                           const float* __restrict__ values,  // (M, C)
                           float* __restrict__ out_keys,      // (T)
                           float* __restrict__ out_vals)      // (T)
{
    __shared__ __align__(16) __bf16 lds_keys[CHUNK_KEYS * C_DIM];  // 32 KB

    const int lane    = threadIdx.x & 31;
    const int wave    = threadIdx.x >> 5;
    const int rowBase = blockIdx.x * ROWS_PER_BLOCK + wave * ROWS_PER_WAVE;

    // ---- B fragments (B = q^T) for RTILES row groups, bf16 in registers ----
    // Per K=32 step f: lane n (0-15) holds column n (q row n), K = f*32 + 0..15
    // contiguous; lane n+16 holds K = f*32 + 16..31.
    const int qcol = lane & 15;
    const int qsel = (lane >> 4) << 4;  // 0 or 16

    v16bf bq[RTILES][4];
#pragma unroll
    for (int rt = 0; rt < RTILES; ++rt) {
        const float* qrow_ptr = q + (size_t)(rowBase + rt * 16 + qcol) * C_DIM;
#pragma unroll
        for (int f = 0; f < 4; ++f) {
            const float4* p = (const float4*)(qrow_ptr + f * 32 + qsel);
#pragma unroll
            for (int g = 0; g < 4; ++g) {
                float4 v = p[g];
                bq[rt][f][4 * g + 0] = (__bf16)v.x;
                bq[rt][f][4 * g + 1] = (__bf16)v.y;
                bq[rt][f][4 * g + 2] = (__bf16)v.z;
                bq[rt][f][4 * g + 3] = (__bf16)v.w;
            }
        }
    }

    // Per accumulator slot: best value + best (uniform) tile id.  Index is
    // reconstructed at the end:  idx = tile*16 + asel + j.
    float bestv[RTILES][8];
    int   bestt[RTILES][8];
#pragma unroll
    for (int rt = 0; rt < RTILES; ++rt)
#pragma unroll
        for (int j = 0; j < 8; ++j) { bestv[rt][j] = -3.402823466e38f; bestt[rt][j] = 0; }

    // A-fragment (keys tile) addressing, matching B's contiguous K order:
    // lane n (0-15): row n, K = f*32+0..15; lane n+16: K = f*32+16..31.
    const int arow = lane & 15;
    const int asel = (lane >> 4) << 3;  // 0 or 8

    for (int cc = 0; cc < N_CHUNKS; ++cc) {
        __syncthreads();  // previous chunk's consumers done before overwrite
        {   // cooperative load + fp32->bf16 convert of 128 keys x 128 ch into LDS
            const float* src = keys + (size_t)cc * CHUNK_KEYS * C_DIM;
            for (int i = threadIdx.x * 4; i < CHUNK_KEYS * C_DIM; i += 256 * 4) {
                float4 v = *(const float4*)(src + i);
                lds_keys[i + 0] = (__bf16)v.x;
                lds_keys[i + 1] = (__bf16)v.y;
                lds_keys[i + 2] = (__bf16)v.z;
                lds_keys[i + 3] = (__bf16)v.w;
            }
        }
        __syncthreads();

#pragma unroll
        for (int t = 0; t < TILES_PER_CHUNK; ++t) {
            const __bf16* krow = &lds_keys[(t * 16 + arow) * C_DIM];
            v8f c[RTILES];
#pragma unroll
            for (int rt = 0; rt < RTILES; ++rt) c[rt] = (v8f){};

            // Each A fragment is loaded once from LDS and reused across both
            // register-resident B tiles -> LDS bytes per WMMA halved.
#pragma unroll
            for (int f = 0; f < 4; ++f) {
                v16bf a;
                *((uint4*)&a + 0) = *(const uint4*)(krow + f * 32 + 2 * asel);
                *((uint4*)&a + 1) = *(const uint4*)(krow + f * 32 + 2 * asel + 8);
#pragma unroll
                for (int rt = 0; rt < RTILES; ++rt) {
                    c[rt] = __builtin_amdgcn_wmma_f32_16x16x32_bf16(
                                false, a, false, bq[rt][f],
                                (short)0, c[rt], false, false);
                }
            }

            // Deferred argmax: 1 cmp + 2 cndmask per score (tile id is SGPR).
            const int tglob = cc * TILES_PER_CHUNK + t;
#pragma unroll
            for (int rt = 0; rt < RTILES; ++rt)
#pragma unroll
                for (int j = 0; j < 8; ++j) {
                    float s = c[rt][j];
                    bool gt = s > bestv[rt][j];
                    bestv[rt][j] = gt ? s : bestv[rt][j];
                    bestt[rt][j] = gt ? tglob : bestt[rt][j];
                }
        }
    }

    // ---- Resolve argmax: per-lane over 8 slots, then across half-lanes ----
    int fidx[RTILES];
#pragma unroll
    for (int rt = 0; rt < RTILES; ++rt) {
        float bv = -3.402823466e38f;
        int   bi = 0x7FFFFFFF;
#pragma unroll
        for (int j = 0; j < 8; ++j) {
            float v   = bestv[rt][j];
            int   idx = bestt[rt][j] * 16 + asel + j;
            if (v > bv || (v == bv && idx < bi)) { bv = v; bi = idx; }
        }
        // Keys interleave by 8 between half-lanes; lower index wins ties.
        float ov = __shfl_xor(bv, 16);
        int   oi = __shfl_xor(bi, 16);
        if (ov > bv || (ov == bv && oi < bi)) { bv = ov; bi = oi; }
        fidx[rt] = bi;  // lane n (0-15) holds argmax for q row rowBase+rt*16+n
    }

    // ---- fp32 squared-distance reductions, one row at a time across the wave ----
#pragma unroll
    for (int rt = 0; rt < RTILES; ++rt) {
        for (int rr = 0; rr < 16; ++rr) {
            const int row = rowBase + rt * 16 + rr;
            const int idx = __shfl(fidx[rt], rr);
            const float4* qv = (const float4*)(q      + (size_t)row * C_DIM);
            const float4* rv = (const float4*)(r      + (size_t)row * C_DIM);
            const float4* kv = (const float4*)(keys   + (size_t)idx * C_DIM);
            const float4* vv = (const float4*)(values + (size_t)idx * C_DIM);
            float4 a4 = qv[lane], b4 = kv[lane];
            float4 c4 = rv[lane], d4 = vv[lane];
            float dk = (a4.x - b4.x) * (a4.x - b4.x) + (a4.y - b4.y) * (a4.y - b4.y)
                     + (a4.z - b4.z) * (a4.z - b4.z) + (a4.w - b4.w) * (a4.w - b4.w);
            float dv = (c4.x - d4.x) * (c4.x - d4.x) + (c4.y - d4.y) * (c4.y - d4.y)
                     + (c4.z - d4.z) * (c4.z - d4.z) + (c4.w - d4.w) * (c4.w - d4.w);
#pragma unroll
            for (int s = 16; s > 0; s >>= 1) {
                dk += __shfl_xor(dk, s);
                dv += __shfl_xor(dv, s);
            }
            if (lane == 0) {
                out_keys[row] = dk;
                out_vals[row] = dv;
            }
        }
    }
}

extern "C" void kernel_launch(void* const* d_in, const int* in_sizes, int n_in,
                              void* d_out, int out_size, void* d_ws, size_t ws_size,
                              hipStream_t stream) {
    const float* trend  = (const float*)d_in[0];  // (32, 2048, 128)
    const float* repr   = (const float*)d_in[1];  // (32, 2048, 128)
    const float* keys   = (const float*)d_in[2];  // (1024, 128)
    const float* values = (const float*)d_in[3];  // (1024, 128)
    float* out = (float*)d_out;                   // [keys_gathering | values_gathering]

    dim3 grid(T_ROWS / ROWS_PER_BLOCK);  // 256 blocks
    dim3 block(256);                     // 8 waves (wave32)
    hipLaunchKernelGGL(gathering_loss_kernel, grid, block, 0, stream,
                       trend, repr, keys, values, out, out + T_ROWS);
}
